// MultiHeadAttention_38113539785306
// MI455X (gfx1250) — matrix operand order repaired and checked by reading, not compiled, not run on hardware
//
#include <hip/hip_runtime.h>
#include <hip/hip_bf16.h>

// ---------------------------------------------------------------------------
// MHA forward for MI455X (gfx1250, wave32, WMMA bf16 w/ f32 accumulate)
// B=1, S=2048, H=1024, NH=16, d_k=64
// bf16 end-to-end; V produced transposed per head so all WMMA B-fragments are
// contiguous LDS b128 loads; K/V^T tiles double-buffered through LDS with
// global_load_async_to_lds_b128 (ASYNCcnt pipeline: DMA overlaps compute).
// ---------------------------------------------------------------------------

typedef __attribute__((ext_vector_type(16))) __bf16 v16bf;
typedef __attribute__((ext_vector_type(8)))  float  v8f;
typedef unsigned short u16;

union Frag {
    v16bf v;
    u16   u[16];
    uint4 q[2];
};

// native f32 -> bf16
__device__ __forceinline__ u16 f2bf(float f) {
    __bf16 h = (__bf16)f;
    return __builtin_bit_cast(u16, h);
}

// ---------------------------------------------------------------------------
// Elementwise f32 -> bf16 (memory-bound, runs once per buffer)
// ---------------------------------------------------------------------------
__global__ __launch_bounds__(256)
void cvt_f32_bf16(const float4* __restrict__ in, u16* __restrict__ out, int n4) {
    const int i = blockIdx.x * 256 + threadIdx.x;
    if (i < n4) {
        float4 f = in[i];
        uint2 p;
        p.x = (unsigned)f2bf(f.x) | ((unsigned)f2bf(f.y) << 16);
        p.y = (unsigned)f2bf(f.z) | ((unsigned)f2bf(f.w) << 16);
        *(uint2*)(out + (size_t)i * 4) = p;
    }
}

// ---------------------------------------------------------------------------
// GEMM: C[M,N] = X[M,K] * W[N,K]^T + bias; X,W bf16; C fp32 or bf16.
// BIAS_ROW=false: bias[n] (normal). BIAS_ROW=true: bias[m] (used for V^T).
// block = 256 threads (8 waves); block tile 128x128; wave tile 32x64.
// ---------------------------------------------------------------------------
template <bool OUT_BF16, bool BIAS_ROW>
__global__ __launch_bounds__(256)
void gemm_bf16(const u16* __restrict__ X, const u16* __restrict__ W,
               const float* __restrict__ bias, void* __restrict__ Cout,
               int M, int N, int K) {
    const int tid   = threadIdx.x;
    const int lane  = tid & 31;
    const int wave  = tid >> 5;      // 0..7
    const int wm    = wave & 3;      // 4 row-slabs of 32
    const int wn    = wave >> 2;     // 2 col-slabs of 64
    const int l16   = lane & 15;
    const int lhalf = (lane >> 4) & 1;

    const int blockM = blockIdx.y * 128;
    const int blockN = blockIdx.x * 128;

    v8f acc[2][4];
#pragma unroll
    for (int i = 0; i < 2; ++i)
#pragma unroll
        for (int j = 0; j < 4; ++j) acc[i][j] = (v8f)(0.0f);

    const int abase = lhalf * 16;    // A half-wave K base (0 / 16)
    const int bbase = lhalf * 16;    // B half-wave K base (0 / 16)

    const u16* xrow[2];
#pragma unroll
    for (int i = 0; i < 2; ++i)
        xrow[i] = X + (size_t)(blockM + wm * 32 + i * 16 + l16) * K + abase;
    const u16* wcol[4];
#pragma unroll
    for (int j = 0; j < 4; ++j)
        wcol[j] = W + (size_t)(blockN + wn * 64 + j * 16 + l16) * K + bbase;

    for (int k0 = 0; k0 < K; k0 += 32) {
        if (k0 + 32 < K) {                 // prefetch next K-slab
            __builtin_prefetch(xrow[0] + k0 + 32, 0, 1);
            __builtin_prefetch(wcol[0] + k0 + 32, 0, 1);
        }
        Frag a[2];
#pragma unroll
        for (int i = 0; i < 2; ++i) {
            const uint4* xp = (const uint4*)(xrow[i] + k0);
            a[i].q[0] = xp[0];             // K = abase .. abase+7
            a[i].q[1] = xp[1];             // K = abase+8 .. abase+15
        }
        Frag b[4];
#pragma unroll
        for (int j = 0; j < 4; ++j) {
            const uint4* wp = (const uint4*)(wcol[j] + k0);
            b[j].q[0] = wp[0];             // K = bbase .. bbase+15
            b[j].q[1] = wp[1];
        }
#pragma unroll
        for (int i = 0; i < 2; ++i)
#pragma unroll
            for (int j = 0; j < 4; ++j)
                acc[i][j] = __builtin_amdgcn_wmma_f32_16x16x32_bf16(
                    false, a[i].v, false, b[j].v, (short)0, acc[i][j], false, false);
    }

    // Epilogue: C layout — VGPR e holds row e (lanes 0-15) / e+8 (lanes 16-31)
#pragma unroll
    for (int i = 0; i < 2; ++i) {
        const int rbase = blockM + wm * 32 + i * 16 + lhalf * 8;
#pragma unroll
        for (int j = 0; j < 4; ++j) {
            const int col = blockN + wn * 64 + j * 16 + l16;
            const float bcol = BIAS_ROW ? 0.0f : bias[col];
#pragma unroll
            for (int e = 0; e < 8; ++e) {
                const float bb = BIAS_ROW ? bias[rbase + e] : bcol;
                const float r = acc[i][j][e] + bb;
                if constexpr (OUT_BF16)
                    ((u16*)Cout)[(size_t)(rbase + e) * N + col] = f2bf(r);
                else
                    ((float*)Cout)[(size_t)(rbase + e) * N + col] = r;
            }
        }
    }
}

// ---------------------------------------------------------------------------
// Streaming (flash-style) attention, one head x 128 query rows per block.
// Q,K bf16 [S,H]; V^T bf16 [H,S]; ctx bf16 [S,H]; fp32 online softmax.
// K / V^T tiles double-buffered via global_load_async_to_lds_b128:
//   top of iter: issue next tile's async loads -> other buffer,
//   s_wait_asynccnt 2 -> current tile complete, DMA overlaps compute.
// Reference quirk preserved: masked scores set to 0 (finite) BEFORE softmax.
// ---------------------------------------------------------------------------
__global__ __launch_bounds__(256)
void attn_kernel(const u16* __restrict__ Q, const u16* __restrict__ Km,
                 const u16* __restrict__ Vt, const int* __restrict__ mask,
                 u16* __restrict__ ctx) {
    constexpr int S  = 2048;
    constexpr int H  = 1024;
    constexpr int DK = 64;
    constexpr int KP = 72;   // K-tile LDS pitch (u16): bank stride 36 -> conflict-free
    constexpr int VP = 40;   // V^T/P LDS pitch (u16): bank stride 20 -> conflict-free

    const int h     = blockIdx.y;
    const int q0    = blockIdx.x * 128;
    const int tid   = threadIdx.x;
    const int lane  = tid & 31;
    const int wave  = tid >> 5;
    const int l16   = lane & 15;
    const int lhalf = (lane >> 4) & 1;
    const int abase = lhalf * 16;

    __shared__ __align__(16) u16 Klds[2][32][KP];    // double-buffered K tile
    __shared__ __align__(16) u16 VldsT[2][64][VP];   // double-buffered V^T tile
    __shared__ __align__(16) u16 Plds[8][16][VP];    // per-wave P patch

    // Persistent Q A-fragments (d_k = 64 -> 2 K-steps of 32), direct bf16 loads
    Frag qa[2];
    {
        const u16* qp = Q + (size_t)(q0 + wave * 16 + l16) * H + h * DK + abase;
#pragma unroll
        for (int ks = 0; ks < 2; ++ks) {
            const uint4* v4 = (const uint4*)(qp + ks * 32);
            qa[ks].q[0] = v4[0];
            qa[ks].q[1] = v4[1];
        }
    }

    v8f oacc[4];
#pragma unroll
    for (int j = 0; j < 4; ++j) oacc[j] = (v8f)(0.0f);
    float mrun[8], lrun[8];
#pragma unroll
    for (int e = 0; e < 8; ++e) { mrun[e] = -1e30f; lrun[e] = 0.0f; }

    const float scale = 0.125f;               // 1/sqrt(64)
    const float L2E   = 1.44269504088896340736f;

    // async staging assignments (one b128 per thread per tile)
    const int skey = tid >> 3, sseg = tid & 7;        // K tile: 32 keys x 8 segs
    const u16* kg = Km + (size_t)skey * H + h * DK + sseg * 8;
    const unsigned kOff0 =
        (unsigned)(uintptr_t)&Klds[0][0][0] + (unsigned)(skey * (KP * 2) + sseg * 16);
    const int sd = tid >> 2, svseg = tid & 3;         // V^T tile: 64 d x 4 segs
    const u16* vgt = Vt + (size_t)(h * DK + sd) * S + svseg * 8;
    const unsigned vOff0 =
        (unsigned)(uintptr_t)&VldsT[0][0][0] + (unsigned)(sd * (VP * 2) + svseg * 16);
    constexpr unsigned KBUF = 32 * KP * 2;   // bytes per K buffer
    constexpr unsigned VBUF = 64 * VP * 2;   // bytes per V buffer

    // per-lane mask row pointer; row offsets e*S + jt*16 are compile-time consts
    const int* mrow = mask + (size_t)(q0 + wave * 16 + lhalf * 8) * S + l16;

    // prologue: async-load tile 0 into buffer 0
    {
        asm volatile("global_load_async_to_lds_b128 %0, %1, off"
                     :: "v"(kOff0), "v"(kg) : "memory");
        asm volatile("global_load_async_to_lds_b128 %0, %1, off"
                     :: "v"(vOff0), "v"(vgt) : "memory");
    }

    for (int j0 = 0; j0 < S; j0 += 32) {
        const int p = (j0 >> 5) & 1;
        // Issue next tile's DMA into the other buffer (wraps harmlessly on last)
        {
            const int jn = (j0 + 32) & (S - 1);
            const u16* gk = kg + (size_t)jn * H;
            const u16* gv = vgt + jn;
            const unsigned kOffN = kOff0 + (p ? 0u : KBUF);
            const unsigned vOffN = vOff0 + (p ? 0u : VBUF);
            asm volatile("global_load_async_to_lds_b128 %0, %1, off"
                         :: "v"(kOffN), "v"(gk) : "memory");
            asm volatile("global_load_async_to_lds_b128 %0, %1, off"
                         :: "v"(vOffN), "v"(gv) : "memory");
        }
        // Wait until only the 2 newest (next-tile) async ops remain in flight:
        // async loads complete in order, so the current tile is now in LDS.
        asm volatile("s_wait_asynccnt 2" ::: "memory");
        __syncthreads();

        const u16 (*Kl)[KP] = Klds[p];
        const u16 (*Vl)[VP] = VldsT[p];

        // Scores: 2 key-subtiles of 16; K-dim = d_k = 64 = 2 WMMA steps
        v8f sc[2];
#pragma unroll
        for (int jt = 0; jt < 2; ++jt) {
            v8f s = (v8f)(0.0f);
#pragma unroll
            for (int ks = 0; ks < 2; ++ks) {
                Frag bf_;
                const u16* kp = &Kl[jt * 16 + l16][ks * 32 + lhalf * 16];
                bf_.q[0] = ((const uint4*)kp)[0];
                bf_.q[1] = ((const uint4*)kp)[1];
                s = __builtin_amdgcn_wmma_f32_16x16x32_bf16(
                    false, qa[ks].v, false, bf_.v, (short)0, s, false, false);
            }
            sc[jt] = s;
        }

        // Scale + faithful mask quirk (masked -> 0, finite).
#pragma unroll
        for (int jt = 0; jt < 2; ++jt) {
#pragma unroll
            for (int e = 0; e < 8; ++e) {
                const float sv = sc[jt][e] * scale;
                const int mv = mrow[e * S + jt * 16];   // const fold -> imm offset
                sc[jt][e] = (mv == 0) ? 0.0f : sv;
            }
        }
        mrow += 32;

        // Online softmax: per-row reductions across the 16 lanes of each half
        float p0e[8], p1e[8];
#pragma unroll
        for (int e = 0; e < 8; ++e) {
            float tmax = fmaxf(sc[0][e], sc[1][e]);
#pragma unroll
            for (int m = 1; m < 16; m <<= 1)
                tmax = fmaxf(tmax, __shfl_xor(tmax, m, 32));
            const float mnew = fmaxf(mrun[e], tmax);
            const float corr = exp2f((mrun[e] - mnew) * L2E);
            const float p0 = exp2f((sc[0][e] - mnew) * L2E);
            const float p1 = exp2f((sc[1][e] - mnew) * L2E);
            float rs = p0 + p1;
#pragma unroll
            for (int m = 1; m < 16; m <<= 1)
                rs += __shfl_xor(rs, m, 32);
            lrun[e] = lrun[e] * corr + rs;
            mrun[e] = mnew;
            p0e[e] = p0; p1e[e] = p1;
#pragma unroll
            for (int j = 0; j < 4; ++j) oacc[j][e] *= corr;
        }

        // C-layout -> A-layout for P via per-wave LDS patch (wave-local: DS ops
        // from one wave are in-order; explicit split-counter wait, no barrier)
#pragma unroll
        for (int e = 0; e < 8; ++e) {
            const int prow = lhalf * 8 + e;
            Plds[wave][prow][l16]      = f2bf(p0e[e]);
            Plds[wave][prow][16 + l16] = f2bf(p1e[e]);
        }
        asm volatile("s_wait_dscnt 0" ::: "memory");
        Frag pa;
        {
            const u16* pp = &Plds[wave][l16][abase];
            pa.q[0] = *(const uint4*)pp;          // k = abase..abase+7
            pa.q[1] = *(const uint4*)(pp + 8);    // k = abase+8..abase+15
        }

        // oacc += P (16x32) * V (32x64): B-fragments contiguous from V^T
#pragma unroll
        for (int j = 0; j < 4; ++j) {
            Frag bv2;
            const u16* vp = &Vl[j * 16 + l16][lhalf * 16];
            bv2.q[0] = ((const uint4*)vp)[0];
            bv2.q[1] = ((const uint4*)vp)[1];
            oacc[j] = __builtin_amdgcn_wmma_f32_16x16x32_bf16(
                false, pa.v, false, bv2.v, (short)0, oacc[j], false, false);
        }
        // all fragment ds_loads have completed (they fed the WMMAs above);
        // barrier before next iteration's async DMA may overwrite this buffer
        __syncthreads();
    }

    // Normalize and store bf16 context in [S,H] (head-interleaved columns)
#pragma unroll
    for (int e = 0; e < 8; ++e) {
        const int qrow = q0 + wave * 16 + lhalf * 8 + e;
        const float inv = 1.0f / lrun[e];
#pragma unroll
        for (int j = 0; j < 4; ++j)
            ctx[(size_t)qrow * H + h * DK + j * 16 + l16] = f2bf(oacc[j][e] * inv);
    }
}

// ---------------------------------------------------------------------------
extern "C" void kernel_launch(void* const* d_in, const int* in_sizes, int n_in,
                              void* d_out, int out_size, void* d_ws, size_t ws_size,
                              hipStream_t stream) {
    (void)in_sizes; (void)n_in; (void)out_size; (void)ws_size;

    const float* q    = (const float*)d_in[0];
    const float* k    = (const float*)d_in[1];
    const float* v    = (const float*)d_in[2];
    const int*   mask = (const int*)  d_in[3];
    const float* Wq = (const float*)d_in[4];  const float* bq = (const float*)d_in[5];
    const float* Wk = (const float*)d_in[6];  const float* bk = (const float*)d_in[7];
    const float* Wv = (const float*)d_in[8];  const float* bv = (const float*)d_in[9];
    const float* Wo = (const float*)d_in[10]; const float* bo = (const float*)d_in[11];
    float* out = (float*)d_out;

    const int S = 2048, H = 1024, NH = 16;
    const size_t SH = (size_t)S * H;   // 2M elements
    const size_t HH = (size_t)H * H;   // 1M elements

    // ws carve-up (all bf16/u16): 7*SH + 4*HH ushorts = 36 MB
    u16* qb   = (u16*)d_ws;
    u16* kb   = qb  + SH;
    u16* vb   = kb  + SH;
    u16* Wqb  = vb  + SH;
    u16* Wkb  = Wqb + HH;
    u16* Wvb  = Wkb + HH;
    u16* Wob  = Wvb + HH;
    u16* Qb   = Wob + HH;
    u16* Kb   = Qb  + SH;
    u16* Vtb  = Kb  + SH;              // V^T per head: [H][S]
    u16* ctxb = Vtb + SH;

    dim3 blk(256);

    // one-time bf16 conversions (memory-bound)
    const int nSH4 = (int)(SH / 4), nHH4 = (int)(HH / 4);
    dim3 cg1((nSH4 + 255) / 256), cg2((nHH4 + 255) / 256);
    cvt_f32_bf16<<<cg1, blk, 0, stream>>>((const float4*)q,  qb,  nSH4);
    cvt_f32_bf16<<<cg1, blk, 0, stream>>>((const float4*)k,  kb,  nSH4);
    cvt_f32_bf16<<<cg1, blk, 0, stream>>>((const float4*)v,  vb,  nSH4);
    cvt_f32_bf16<<<cg2, blk, 0, stream>>>((const float4*)Wq, Wqb, nHH4);
    cvt_f32_bf16<<<cg2, blk, 0, stream>>>((const float4*)Wk, Wkb, nHH4);
    cvt_f32_bf16<<<cg2, blk, 0, stream>>>((const float4*)Wv, Wvb, nHH4);
    cvt_f32_bf16<<<cg2, blk, 0, stream>>>((const float4*)Wo, Wob, nHH4);

    dim3 ggrid(H / 128, S / 128);      // (N tiles, M tiles)
    gemm_bf16<true , false><<<ggrid, blk, 0, stream>>>(qb, Wqb, bq, Qb, S, H, H);
    gemm_bf16<true , false><<<ggrid, blk, 0, stream>>>(kb, Wkb, bk, Kb, S, H, H);
    // V^T[channel][seq] = Wv @ v^T : same kernel with swapped operand roles,
    // bias applied per output ROW (= channel)
    dim3 vgrid(S / 128, H / 128);
    gemm_bf16<true , true ><<<vgrid, blk, 0, stream>>>(Wvb, vb, bv, Vtb, H, S, H);

    dim3 agrid(S / 128, NH);
    attn_kernel<<<agrid, blk, 0, stream>>>(Qb, Kb, Vtb, mask, ctxb);

    gemm_bf16<false, false><<<ggrid, blk, 0, stream>>>(ctxb, Wob, bo, out, S, H, H);
}
